// GatingNetwork_35338990912075
// MI455X (gfx1250) — matrix operand order repaired and checked by reading, not compiled, not run on hardware
//
#include <hip/hip_runtime.h>
#include <hip/hip_bf16.h>

typedef __bf16 bf16_t;
typedef bf16_t v16bf __attribute__((ext_vector_type(16)));
typedef float  v8f   __attribute__((ext_vector_type(8)));

#define B_TOK 131072
#define DDIM  1024
#define NEXP  64
#define EPS_F 1e-12f

// ---------------------------------------------------------------------------
// Kernel 0: pack W [64 experts x 1024] f32 into the WMMA B-operand bf16 layout.
// B tile for (k-chunk kc, n-tile t): lane l holds column N = t*16 + (l&15), with the
// 16 bf16 values K = kc*32 + 8*(l>>4) + j (j<8), kc*32 + 16 + 8*(l>>4) + j-8 (j>=8).
// Bpack element offset = ((kc*4 + t)*32 + lane)*16.  Total: 128 KB in d_ws.
// ---------------------------------------------------------------------------
__global__ __launch_bounds__(256) void pack_W_kernel(const float* __restrict__ W,
                                                     bf16_t* __restrict__ Bp) {
    int g = blockIdx.x * blockDim.x + threadIdx.x;   // 0 .. 4095
    if (g >= 32 * 4 * 32) return;
    int lane = g & 31;
    int t    = (g >> 5) & 3;
    int kc   = g >> 7;
    int n    = t * 16 + (lane & 15);                  // expert index
    bf16_t* dst = Bp + (size_t)g * 16;
#pragma unroll
    for (int j = 0; j < 16; ++j) {
        int k = kc * 32 + ((j < 8) ? (lane >> 4) * 8 + j : 16 + (lane >> 4) * 8 + (j - 8));
        dst[j] = (bf16_t)W[(size_t)n * DDIM + k];
    }
}

// ---------------------------------------------------------------------------
// Kernel 1: fused router GEMM (bf16 WMMA) + softmax + top-8 + top-2.
// Block = 256 threads = 8 waves; each wave owns 16 tokens x 64 experts.
// ---------------------------------------------------------------------------
__global__ __launch_bounds__(256) void gating_kernel(const float* __restrict__ x,
                                                     const bf16_t* __restrict__ Bp,
                                                     long long* __restrict__ topk_idx,
                                                     float*     __restrict__ topk_w,
                                                     float*     __restrict__ probs,
                                                     long long* __restrict__ topc_idx) {
    __shared__ float lds[128 * 68];                  // 128 tokens, stride 68 (pad)

    const int lane = threadIdx.x & 31;
    const int wave = threadIdx.x >> 5;
    const int row  = lane & 15;                      // A-matrix M within tile
    const int half = lane >> 4;                      // which K-half this lane loads

    const int tok = blockIdx.x * 128 + wave * 16 + row;
    const float* __restrict__ xrow = x + (size_t)tok * DDIM;

    v8f acc0 = {}, acc1 = {}, acc2 = {}, acc3 = {};

    for (int kc = 0; kc < 32; ++kc) {
        // ---- A operand: 16 f32 of this token's row -> bf16, WMMA A layout.
        // lanes 0-15: K = k0+0..7 (elems 0..7), k0+16..23 (elems 8..15)
        // lanes16-31: K = k0+8..15,              k0+24..31
        const int kA = kc * 32 + half * 8;
        float4 f0 = *(const float4*)(xrow + kA);
        float4 f1 = *(const float4*)(xrow + kA + 4);
        float4 f2 = *(const float4*)(xrow + kA + 16);
        float4 f3 = *(const float4*)(xrow + kA + 20);
        v16bf a;
        a[0]  = (bf16_t)f0.x; a[1]  = (bf16_t)f0.y; a[2]  = (bf16_t)f0.z; a[3]  = (bf16_t)f0.w;
        a[4]  = (bf16_t)f1.x; a[5]  = (bf16_t)f1.y; a[6]  = (bf16_t)f1.z; a[7]  = (bf16_t)f1.w;
        a[8]  = (bf16_t)f2.x; a[9]  = (bf16_t)f2.y; a[10] = (bf16_t)f2.z; a[11] = (bf16_t)f2.w;
        a[12] = (bf16_t)f3.x; a[13] = (bf16_t)f3.y; a[14] = (bf16_t)f3.z; a[15] = (bf16_t)f3.w;

        // ---- B operand: prepacked, L2-resident; 32-byte coalesced loads.
        const v16bf* bp = ((const v16bf*)Bp) + (size_t)(kc * 4) * 32 + lane;
        v16bf b0 = bp[0];
        v16bf b1 = bp[32];
        v16bf b2 = bp[64];
        v16bf b3 = bp[96];

        acc0 = __builtin_amdgcn_wmma_f32_16x16x32_bf16(false, a, false, b0, (short)0, acc0, false, false);
        acc1 = __builtin_amdgcn_wmma_f32_16x16x32_bf16(false, a, false, b1, (short)0, acc1, false, false);
        acc2 = __builtin_amdgcn_wmma_f32_16x16x32_bf16(false, a, false, b2, (short)0, acc2, false, false);
        acc3 = __builtin_amdgcn_wmma_f32_16x16x32_bf16(false, a, false, b3, (short)0, acc3, false, false);
    }

    // ---- Spill logits to LDS. Acc layout: VGPR r = row M=r (lanes 0-15) or
    // M=r+8 (lanes 16-31); N = (lane&15) + 16*tile.  TAU = 1.0 (folded).
    {
        const int lrow = wave * 16 + 8 * half;
#pragma unroll
        for (int r = 0; r < 8; ++r) {
            float* dst = &lds[(lrow + r) * 68 + row];
            dst[0]  = acc0[r];
            dst[16] = acc1[r];
            dst[32] = acc2[r];
            dst[48] = acc3[r];
        }
    }
    __syncthreads();

    // ---- Per-token softmax + top-8 + top-2 (one thread per token).
    if (threadIdx.x < 128) {
        const int gt = blockIdx.x * 128 + (int)threadIdx.x;
        const float* vrow = &lds[threadIdx.x * 68];

        float v[NEXP];
#pragma unroll
        for (int i = 0; i < 16; ++i) {
            float4 q = *(const float4*)(vrow + i * 4);
            v[i * 4 + 0] = q.x; v[i * 4 + 1] = q.y;
            v[i * 4 + 2] = q.z; v[i * 4 + 3] = q.w;
        }

        float vmax = v[0];
#pragma unroll
        for (int i = 1; i < NEXP; ++i) vmax = fmaxf(vmax, v[i]);

        float sum = 0.0f;
#pragma unroll
        for (int i = 0; i < NEXP; ++i) sum += __expf(v[i] - vmax);
        const float inv = 1.0f / (sum + EPS_F);

        float4* pout = (float4*)(probs + (size_t)gt * NEXP);
#pragma unroll
        for (int i = 0; i < 16; ++i) {
            float4 q;
            q.x = __expf(v[i * 4 + 0] - vmax) * inv;
            q.y = __expf(v[i * 4 + 1] - vmax) * inv;
            q.z = __expf(v[i * 4 + 2] - vmax) * inv;
            q.w = __expf(v[i * 4 + 3] - vmax) * inv;
            pout[i] = q;
        }

        // top-8 selection: strict '>' scanning ascending i => lowest-index ties,
        // matching jax.lax.top_k; results come out sorted descending.
        unsigned long long chosen = 0ull;
        float cv[8];
        int   ci[8];
        for (int s = 0; s < 8; ++s) {
            float best = -3.402823466e38f;
            int   bi   = 0;
#pragma unroll
            for (int i = 0; i < NEXP; ++i) {
                bool used = (chosen >> i) & 1ull;
                if (!used && v[i] > best) { best = v[i]; bi = i; }
            }
            chosen |= (1ull << bi);
            cv[s] = best;
            ci[s] = bi;
        }

        long long* tci = topc_idx + (size_t)gt * 8;
#pragma unroll
        for (int s = 0; s < 8; ++s) tci[s] = (long long)ci[s];

        // top-2 of the (already sorted) top-8 => first two entries.
        const float m2 = fmaxf(cv[0], cv[1]);
        const float e0 = __expf(cv[0] - m2);
        const float e1 = __expf(cv[1] - m2);
        const float is = 1.0f / (e0 + e1 + EPS_F);

        topk_idx[(size_t)gt * 2 + 0] = (long long)ci[0];
        topk_idx[(size_t)gt * 2 + 1] = (long long)ci[1];
        topk_w  [(size_t)gt * 2 + 0] = e0 * is;
        topk_w  [(size_t)gt * 2 + 1] = e1 * is;
    }
}

// ---------------------------------------------------------------------------
// Host-side launcher. Outputs concatenated flat in return order:
//   topk_idx  int64 [B,2]   @ byte 0
//   topk_w    f32   [B,2]   @ B*16
//   probs     f32   [B,64]  @ B*16 + B*8
//   topc_idx  int64 [B,8]   @ B*16 + B*8 + B*256
// ---------------------------------------------------------------------------
extern "C" void kernel_launch(void* const* d_in, const int* in_sizes, int n_in,
                              void* d_out, int out_size, void* d_ws, size_t ws_size,
                              hipStream_t stream) {
    (void)in_sizes; (void)n_in; (void)out_size; (void)ws_size;

    const float* x = (const float*)d_in[0];   // [B, D] f32
    const float* W = (const float*)d_in[1];   // [M, D] f32

    bf16_t* Bp = (bf16_t*)d_ws;               // 128 KB prepacked weights

    char* out = (char*)d_out;
    long long* topk_idx = (long long*)(out);
    float*     topk_w   = (float*)(out + (size_t)B_TOK * 2 * 8);
    float*     probs    = (float*)(out + (size_t)B_TOK * 2 * 8 + (size_t)B_TOK * 2 * 4);
    long long* topc_idx = (long long*)(out + (size_t)B_TOK * 2 * 8 + (size_t)B_TOK * 2 * 4 +
                                       (size_t)B_TOK * 64 * 4);

    pack_W_kernel<<<16, 256, 0, stream>>>(W, Bp);
    gating_kernel<<<B_TOK / 128, 256, 0, stream>>>(x, Bp, topk_idx, topk_w, probs, topc_idx);
}
